// Head_81767587381395
// MI455X (gfx1250) — matrix operand order repaired and checked by reading, not compiled, not run on hardware
//
#include <hip/hip_runtime.h>
#include <math.h>

#define B_ 4
#define S_ 4096
#define D_ 1024
#define H_ 128

typedef __attribute__((ext_vector_type(16))) __bf16       v16bf;
typedef __attribute__((ext_vector_type(8)))  float        v8f;
typedef __attribute__((ext_vector_type(4)))  float        v4f;
typedef __attribute__((ext_vector_type(4)))  unsigned int v4u;
typedef __attribute__((ext_vector_type(8)))  unsigned int v8u;
typedef __attribute__((ext_vector_type(4)))  unsigned int u32x4;
typedef __attribute__((ext_vector_type(4)))  int          i32x4;
typedef __attribute__((ext_vector_type(8)))  int          i32x8;

union BF16V {
    v16bf bf;
    v8u   u;
    v4u   q[2];
};

__device__ inline v16bf load_bf16x16_contig(const __bf16* p) {
    BF16V t;
    t.u = *(const v8u*)p;           // 32B contiguous
    return t.bf;
}

__device__ inline v16bf load_bf16x16_2x16(const __bf16* p0, const __bf16* p1) {
    BF16V t;
    t.q[0] = *(const v4u*)p0;       // 16B
    t.q[1] = *(const v4u*)p1;       // 16B
    return t.bf;
}

__device__ inline v8f wmma_bf16(v16bf a, v16bf b, v8f c) {
    // D = A(16x32 bf16) * B(32x16 bf16) + C(16x16 f32)
    return __builtin_amdgcn_wmma_f32_16x16x32_bf16(
        false, a, false, b, (short)0, c, false, false);
}

// ---------------------------------------------------------------------------
// TDM: 2D tile (bf16 elements) global -> LDS. D# per CDNA5 ISA.
//   group0: count=1 | lds_addr | global_addr(57b) | type=2
//   group1: data_size=1(2B), tensor_dim0/1, tile_dim0/1, tensor_dim0_stride
// Dim0 is the contiguous (innermost) dimension, sizes in elements.
// ---------------------------------------------------------------------------
__device__ inline void tdm_load_2d_bf16(unsigned lds_off_bytes,
                                        const __bf16* gptr,
                                        unsigned tile_d0, unsigned tile_d1,
                                        unsigned tensor_d0, unsigned tensor_d1,
                                        unsigned long long stride_d0_elems) {
    unsigned long long ga = (unsigned long long)(uintptr_t)gptr;
    u32x4 g0;
    g0[0] = 1u;                                   // count=1, user mode
    g0[1] = lds_off_bytes;                        // lds_addr (bytes)
    g0[2] = (unsigned)ga;                         // global_addr[31:0]
    g0[3] = (unsigned)((ga >> 32) & 0x01FFFFFFu)  // global_addr[56:32]
          | (2u << 30);                           // type=2 ("image")
    unsigned w0 = (1u << 16);                     // workgroup_mask=0, data_size=1 (2B)
    unsigned w1 = (tensor_d0 & 0xFFFFu) << 16;    // tensor_dim0[15:0] @ bits63:48
    unsigned w2 = (tensor_d0 >> 16) | ((tensor_d1 & 0xFFFFu) << 16);
    unsigned w3 = (tensor_d1 >> 16) | (tile_d0 << 16);   // tile_dim0 @ 127:112
    unsigned w4 = tile_d1;                        // tile_dim1; tile_dim2=0
    unsigned w5 = (unsigned)(stride_d0_elems & 0xFFFFFFFFull);
    unsigned w6 = (unsigned)((stride_d0_elems >> 32) & 0xFFFFull); // dim1_stride=0
    unsigned w7 = 0u;
    i32x8 g1;
    g1[0] = (int)w0; g1[1] = (int)w1; g1[2] = (int)w2; g1[3] = (int)w3;
    g1[4] = (int)w4; g1[5] = (int)w5; g1[6] = (int)w6; g1[7] = (int)w7;
    i32x4 z4 = {0, 0, 0, 0};
#if defined(__clang_major__) && (__clang_major__ >= 23)
    i32x8 z8 = {0, 0, 0, 0, 0, 0, 0, 0};
    __builtin_amdgcn_tensor_load_to_lds(g0, g1, z4, z4, z8, 0);
#else
    __builtin_amdgcn_tensor_load_to_lds(g0, g1, z4, z4, 0);
#endif
}

__device__ inline unsigned lds_offset_of(const void* p) {
    // addrspace(3) offsets live in the low 32 bits of the generic address
    return (unsigned)(uintptr_t)p;
}

// ---------------------------------------------------------------------------
// Kernel 0: W[H][D] f32 -> WT[D][H] bf16
// ---------------------------------------------------------------------------
__global__ __launch_bounds__(256) void wt_kernel(const float* __restrict__ W,
                                                 __bf16* __restrict__ WT) {
    int i = blockIdx.x * 256 + threadIdx.x;    // i in [0, H_*D_)
    int h = i / D_;
    int d = i - h * D_;
    WT[(size_t)d * H_ + h] = (__bf16)W[i];
}

// ---------------------------------------------------------------------------
// Kernel 1: projection. Each wave: 16(s) x 128(h) stripe of one of q/k/v.
// The 4 waves of a block share the B operand (WT d-chunk slab), so wave 0
// stages it via TDM into LDS, double-buffered and overlapped with compute.
// ---------------------------------------------------------------------------
__global__ __launch_bounds__(128) void proj_kernel(
    const float*  __restrict__ X,     // [B,S,D] f32
    const __bf16* __restrict__ WT,    // [D,H]  bf16
    const float*  __restrict__ bias,  // [H]
    __bf16*       __restrict__ outRM, // [B,S,H] or null
    __bf16*       __restrict__ outT)  // [B,H,S] or null
{
    __shared__ __bf16 wtbuf[2][32 * H_];   // [d_local][h] 8KB x2
    __shared__ __bf16 lds[4][H_ * 16];     // per-wave 128h x 16s tile (4KB)

    int wid  = threadIdx.x >> 5;
    int lane = threadIdx.x & 31;
    int tile = blockIdx.x * 4 + wid;     // 0..1023
    int b    = tile >> 8;
    int s0   = (tile & 255) << 4;

    const float* Xb = X + ((size_t)b * S_ + s0) * D_;
    int m    = lane & 15;
    int half = lane >> 4;

    v8f acc[8];
#pragma unroll
    for (int t = 0; t < 8; ++t) acc[t] = (v8f){0.f,0.f,0.f,0.f,0.f,0.f,0.f,0.f};

    // Prologue: stage WT rows [0,32) into buffer 0.
    if (wid == 0) {
        tdm_load_2d_bf16(lds_offset_of(&wtbuf[0][0]), WT,
                         /*tile_d0=*/H_, /*tile_d1=*/32,
                         /*tensor_d0=*/H_, /*tensor_d1=*/D_,
                         /*stride_d0=*/(unsigned long long)H_);
    }

    const float* rowbase = Xb + (size_t)m * D_ + half * 8;
#pragma unroll 1
    for (int it = 0; it < D_ / 32; ++it) {
        int d0  = it * 32;
        int cur = it & 1;

        if (wid == 0) {
            if (it + 1 < D_ / 32) {
                tdm_load_2d_bf16(lds_offset_of(&wtbuf[cur ^ 1][0]),
                                 WT + (size_t)(d0 + 32) * H_,
                                 H_, 32, H_, D_, (unsigned long long)H_);
                __builtin_amdgcn_s_wait_tensorcnt(1);
            } else {
                __builtin_amdgcn_s_wait_tensorcnt(0);
            }
        }
        __syncthreads();   // publish WT slab

        // A tile (16x32): lane=row m, elems 0..7 -> K=half*8+e, 8..15 -> +16
        v4f f0 = *(const v4f*)(rowbase + d0);
        v4f f1 = *(const v4f*)(rowbase + d0 + 4);
        v4f f2 = *(const v4f*)(rowbase + d0 + 16);
        v4f f3 = *(const v4f*)(rowbase + d0 + 20);
        v16bf a;
#pragma unroll
        for (int i = 0; i < 4; ++i) {
            a[i]      = (__bf16)f0[i];
            a[4 + i]  = (__bf16)f1[i];
            a[8 + i]  = (__bf16)f2[i];
            a[12 + i] = (__bf16)f3[i];
        }
        // B tiles (32x16): lane = col h (lane&15), elem s = K-row in table order
        const __bf16* wp = &wtbuf[cur][lane & 15];
#pragma unroll
        for (int ht = 0; ht < 8; ++ht) {
            v16bf bt;
#pragma unroll
            for (int s = 0; s < 16; ++s) {
                int k = (s < 8) ? half * 8 + s : 16 + half * 8 + (s - 8);
                bt[s] = wp[k * H_ + ht * 16];
            }
            acc[ht]  = wmma_bf16(a, bt, acc[ht]);
        }
        __syncthreads();   // all waves done with wtbuf[cur] before TDM reuse
    }

    int ncol = lane & 15;
    if (outRM) {
        __bf16* orow = outRM + ((size_t)b * S_ + s0) * H_;
#pragma unroll
        for (int ht = 0; ht < 8; ++ht) {
            float bv = bias[ht * 16 + ncol];
#pragma unroll
            for (int r = 0; r < 8; ++r) {
                float val = acc[ht][r] + bv;
                orow[(size_t)(r + 8 * half) * H_ + ht * 16 + ncol] = (__bf16)val;
            }
        }
    } else {
        __bf16* L = lds[wid];
#pragma unroll
        for (int ht = 0; ht < 8; ++ht) {
            float bv = bias[ht * 16 + ncol];
#pragma unroll
            for (int r = 0; r < 8; ++r) {
                L[(ht * 16 + ncol) * 16 + r + 8 * half] = (__bf16)(acc[ht][r] + bv);
            }
        }
        asm volatile("s_wait_dscnt 0x0" ::: "memory");
#pragma unroll
        for (int j = 0; j < 4; ++j) {
            int h = j * 32 + lane;
            const v4u* src = (const v4u*)&L[h * 16];
            __bf16* dst = outT + ((size_t)b * H_ + h) * S_ + s0;
            *(v4u*)dst       = src[0];
            *(v4u*)(dst + 8) = src[1];
        }
    }
}

// ---------------------------------------------------------------------------
// Kernel 2: flash attention with TDM-staged K/V tiles.
// Block = 8 waves = 8 query tiles (same batch). Key block = 64.
// ---------------------------------------------------------------------------
#define KB_ 64
#define NKB_ (S_ / KB_)

__global__ __launch_bounds__(256) void attn_kernel(
    const __bf16* __restrict__ qm,    // [B,S,H] bf16
    const __bf16* __restrict__ kT,    // [B,H,S] bf16
    const __bf16* __restrict__ vm,    // [B,S,H] bf16
    const int*    __restrict__ pmask, // [B,S]
    float*        __restrict__ out)   // [B,S,H] f32
{
    __shared__ __bf16 kbuf[2][H_ * KB_];   // [h][key]  16KB x2
    __shared__ __bf16 vbuf[2][KB_ * H_];   // [key][h]  16KB x2
    __shared__ __bf16 Pbuf[8][16 * KB_];   // per-wave P tile 2KB x8

    int wid  = threadIdx.x >> 5;
    int lane = threadIdx.x & 31;
    int tile = blockIdx.x * 8 + wid;       // 0..1023 (8 tiles share a batch)
    int b    = tile >> 8;
    int s0   = (tile & 255) << 4;
    int m    = lane & 15;
    int half = lane >> 4;
    const float scale = 0.08838834764831844f;   // 1/sqrt(128)

    const __bf16* kTb = kT + (size_t)b * H_ * S_;
    const __bf16* vb  = vm + (size_t)b * S_ * H_;
    const int*    pmb = pmask + (size_t)b * S_;

    // Preload q A-tiles (16 queries x 128 head dims = 4 tiles of 16x32)
    v16bf aq[4];
    {
        const __bf16* qrow = qm + ((size_t)b * S_ + s0 + m) * H_ + half * 8;
#pragma unroll
        for (int c = 0; c < 4; ++c)
            aq[c] = load_bf16x16_2x16(qrow + c * 32, qrow + c * 32 + 16);
    }

    v8f o[8];
#pragma unroll
    for (int t = 0; t < 8; ++t) o[t] = (v8f){0.f,0.f,0.f,0.f,0.f,0.f,0.f,0.f};
    float mrow[8], lrow[8];
#pragma unroll
    for (int r = 0; r < 8; ++r) { mrow[r] = -3.0e38f; lrow[r] = 0.f; }

    __bf16* Pw = Pbuf[wid];

    // Prologue: TDM fetch of key-block 0 into buffer 0 (wave 0 only).
    if (wid == 0) {
        tdm_load_2d_bf16(lds_offset_of(&kbuf[0][0]), kTb,
                         /*tile_d0=*/KB_, /*tile_d1=*/H_,
                         /*tensor_d0=*/S_, /*tensor_d1=*/H_,
                         /*stride_d0=*/(unsigned long long)S_);
        tdm_load_2d_bf16(lds_offset_of(&vbuf[0][0]), vb,
                         /*tile_d0=*/H_, /*tile_d1=*/KB_,
                         /*tensor_d0=*/H_, /*tensor_d1=*/S_,
                         /*stride_d0=*/(unsigned long long)H_);
    }

#pragma unroll 1
    for (int it = 0; it < NKB_; ++it) {
        int kb  = it * KB_;
        int cur = it & 1;

        if (wid == 0) {
            if (it + 1 < NKB_) {
                int nb = kb + KB_;
                tdm_load_2d_bf16(lds_offset_of(&kbuf[cur ^ 1][0]), kTb + nb,
                                 KB_, H_, S_, H_, (unsigned long long)S_);
                tdm_load_2d_bf16(lds_offset_of(&vbuf[cur ^ 1][0]),
                                 vb + (size_t)nb * H_,
                                 H_, KB_, H_, S_, (unsigned long long)H_);
                __builtin_amdgcn_s_wait_tensorcnt(2);
            } else {
                __builtin_amdgcn_s_wait_tensorcnt(0);
            }
        }
        __syncthreads();   // publish staged K/V tile to all waves

        // ---- scores: four 16x16 tiles (keys kb+16*kt), K-dim = H over LDS ----
        v8f st[4];
#pragma unroll
        for (int kt = 0; kt < 4; ++kt)
            st[kt] = (v8f){0.f,0.f,0.f,0.f,0.f,0.f,0.f,0.f};
#pragma unroll
        for (int c = 0; c < 4; ++c) {
            const __bf16* kp = &kbuf[cur][(c * 32) * KB_ + m];
#pragma unroll
            for (int kt = 0; kt < 4; ++kt) {
                v16bf bk;
#pragma unroll
                for (int s = 0; s < 16; ++s) {
                    int k = (s < 8) ? half * 8 + s : 16 + half * 8 + (s - 8);
                    bk[s] = kp[k * KB_ + kt * 16];
                }
                st[kt] = wmma_bf16(aq[c], bk, st[kt]);
            }
        }

        int msk[4];
#pragma unroll
        for (int kt = 0; kt < 4; ++kt) msk[kt] = pmb[kb + kt * 16 + m];

        // ---- online softmax (rows = r + 8*half, reduce across 16 lanes) ----
#pragma unroll
        for (int r = 0; r < 8; ++r) {
            float a0 = msk[0] ? st[0][r] : -1e9f;
            float a1 = msk[1] ? st[1][r] : -1e9f;
            float a2 = msk[2] ? st[2][r] : -1e9f;
            float a3 = msk[3] ? st[3][r] : -1e9f;
            float mx = fmaxf(fmaxf(a0, a1), fmaxf(a2, a3));
            mx = fmaxf(mx, __shfl_xor(mx, 1, 32));
            mx = fmaxf(mx, __shfl_xor(mx, 2, 32));
            mx = fmaxf(mx, __shfl_xor(mx, 4, 32));
            mx = fmaxf(mx, __shfl_xor(mx, 8, 32));
            float mnew  = fmaxf(mrow[r], mx);
            float alpha = __expf((mrow[r] - mnew) * scale);
            mrow[r] = mnew;
            float p0 = __expf((a0 - mnew) * scale);
            float p1 = __expf((a1 - mnew) * scale);
            float p2 = __expf((a2 - mnew) * scale);
            float p3 = __expf((a3 - mnew) * scale);
            float rs = (p0 + p1) + (p2 + p3);
            rs += __shfl_xor(rs, 1, 32);
            rs += __shfl_xor(rs, 2, 32);
            rs += __shfl_xor(rs, 4, 32);
            rs += __shfl_xor(rs, 8, 32);
            lrow[r] = lrow[r] * alpha + rs;
#pragma unroll
            for (int ht = 0; ht < 8; ++ht) o[ht][r] *= alpha;
            int rowm = r + 8 * half;
            Pw[rowm * KB_ +  0 + m] = (__bf16)p0;
            Pw[rowm * KB_ + 16 + m] = (__bf16)p1;
            Pw[rowm * KB_ + 32 + m] = (__bf16)p2;
            Pw[rowm * KB_ + 48 + m] = (__bf16)p3;
        }
        asm volatile("s_wait_dscnt 0x0" ::: "memory");

        // ---- reload P as two WMMA A operands (16x32 each) ----
        v16bf ap[2];
#pragma unroll
        for (int ks = 0; ks < 2; ++ks) {
            const __bf16* pp = &Pw[m * KB_ + ks * 32 + half * 8];
            ap[ks] = load_bf16x16_2x16(pp, pp + 16);
        }

        // ---- O += P @ V  (V from LDS) ----
#pragma unroll
        for (int ks = 0; ks < 2; ++ks) {
#pragma unroll
            for (int ht = 0; ht < 8; ++ht) {
                const __bf16* vp = &vbuf[cur][(ks * 32) * H_ + ht * 16 + m];
                v16bf btv;
#pragma unroll
                for (int s = 0; s < 16; ++s) {
                    int k = (s < 8) ? half * 8 + s : 16 + half * 8 + (s - 8);
                    btv[s] = vp[k * H_];
                }
                o[ht] = wmma_bf16(ap[ks], btv, o[ht]);
            }
        }
        __syncthreads();   // all waves done with buf[cur] before TDM reuses it
    }

    // ---- normalize & write f32 output ----
    float* orow = out + ((size_t)b * S_ + s0) * H_;
#pragma unroll
    for (int r = 0; r < 8; ++r) {
        float inv = lrow[r] > 0.f ? 1.f / lrow[r] : 0.f;
#pragma unroll
        for (int ht = 0; ht < 8; ++ht) {
            orow[(size_t)(r + 8 * half) * H_ + ht * 16 + m] = o[ht][r] * inv;
        }
    }
}

// ---------------------------------------------------------------------------
extern "C" void kernel_launch(void* const* d_in, const int* in_sizes, int n_in,
                              void* d_out, int out_size, void* d_ws, size_t ws_size,
                              hipStream_t stream) {
    const float* x  = (const float*)d_in[0];
    const float* Wk = (const float*)d_in[1];
    const float* bk = (const float*)d_in[2];
    const float* Wq = (const float*)d_in[3];
    const float* bq = (const float*)d_in[4];
    const float* Wv = (const float*)d_in[5];
    const float* bv = (const float*)d_in[6];
    const int*   pm = (const int*)d_in[7];
    float* out = (float*)d_out;

    char* ws = (char*)d_ws;
    __bf16* WTk = (__bf16*)ws; ws += (size_t)D_ * H_ * 2;
    __bf16* WTq = (__bf16*)ws; ws += (size_t)D_ * H_ * 2;
    __bf16* WTv = (__bf16*)ws; ws += (size_t)D_ * H_ * 2;
    __bf16* qb  = (__bf16*)ws; ws += (size_t)B_ * S_ * H_ * 2;
    __bf16* kTb = (__bf16*)ws; ws += (size_t)B_ * S_ * H_ * 2;
    __bf16* vbf = (__bf16*)ws; ws += (size_t)B_ * S_ * H_ * 2;

    // Stage 0: weight transpose + bf16 convert
    wt_kernel<<<(H_ * D_) / 256, 256, 0, stream>>>(Wk, WTk);
    wt_kernel<<<(H_ * D_) / 256, 256, 0, stream>>>(Wq, WTq);
    wt_kernel<<<(H_ * D_) / 256, 256, 0, stream>>>(Wv, WTv);

    // Stage 1: projections (1024 s-tiles, 4 waves per block, TDM-staged WT)
    proj_kernel<<<256, 128, 0, stream>>>(x, WTk, bk, nullptr, kTb);  // K -> [B,H,S]
    proj_kernel<<<256, 128, 0, stream>>>(x, WTq, bq, qb, nullptr);   // Q -> [B,S,H]
    proj_kernel<<<256, 128, 0, stream>>>(x, WTv, bv, vbf, nullptr);  // V -> [B,S,H]

    // Stage 2: fused flash attention with TDM double-buffered K/V staging
    attn_kernel<<<128, 256, 0, stream>>>(qb, kTb, vbf, pm, out);
}
